// MPNN_38208029065464
// MI455X (gfx1250) — matrix operand order repaired and checked by reading, not compiled, not run on hardware
//
#include <hip/hip_runtime.h>
#include <hip/hip_bf16.h>

// ---------------------------------------------------------------------------
// Types for CDNA5 WMMA
// ---------------------------------------------------------------------------
typedef __attribute__((ext_vector_type(16))) __bf16 v16bf;
typedef __attribute__((ext_vector_type(8)))  float  v8f;

union AFrag {
    v16bf v;
    uint4 q[2];
};

__device__ __forceinline__ unsigned short f2bf(float f) {
    unsigned int u = __float_as_uint(f);
    unsigned int lsb = (u >> 16) & 1u;
    u += 0x7FFFu + lsb;                // round-to-nearest-even
    return (unsigned short)(u >> 16);
}

__device__ __forceinline__ unsigned int pack2bf(float lo, float hi) {
    return (unsigned int)f2bf(lo) | ((unsigned int)f2bf(hi) << 16);
}

__device__ __forceinline__ float lrelu(float v) {
    return v > 0.0f ? v : 0.1f * v;
}

// ---------------------------------------------------------------------------
// Generic GEMM:  C[n,m] = act( sum_k A[n,k] * W[k,m] + bias[m] )
// REQUIREMENTS (all call sites obey): K % 32 == 0, M % 128 == 0, A fully
// readable for all k in [0,K) (x_ft is pre-padded to 64 cols).
// Out-of-range rows are handled by CLAMPING the load row (results for
// clamped rows are never stored), so the K-loop body is branch-free:
//   A tile: 4x global_load_b128 -> bf16 pack -> 2x ds_store_b128
//   W tile: 4x global_load_b128 -> 16x ds_store_b16 transpose
//   12x ds_load_b128 fragment reads -> 8x v_wmma_f32_16x16x32_bf16
// Tile: 128 rows x 128 cols, K-step 32. 256 threads = 8 waves in a 4x2
// grid; each wave computes a 32x64 region via 2x4 WMMA (A frags reused
// across 4 column sub-tiles, B frags across 2 row sub-tiles).
// ---------------------------------------------------------------------------
__global__ __launch_bounds__(256)
void gemm_bias_act_kernel(const float* __restrict__ A, int lda,
                          const float* __restrict__ W, int ldw,
                          const float* __restrict__ bias,
                          float* __restrict__ C, int ldc,
                          int nRows, int K, int M, int doAct)
{
    __shared__ __align__(16) unsigned short As[128][32];   // bf16 bits, 8 KB
    __shared__ __align__(16) unsigned short Bs[128][32];   // transposed [n][k], 8 KB

    const int tid    = threadIdx.x;
    const int wid    = tid >> 5;
    const int lane   = tid & 31;
    const int laneHi = (lane >> 4) & 1;
    const int lane15 = lane & 15;

    const int rowBase = blockIdx.y * 128;
    const int colBase = blockIdx.x * 128;
    const bool fullRows = (rowBase + 128) <= nRows;

    // A staging: thread -> (row, 16-wide k half); clamp row so loads are
    // always in-bounds and unconditional.
    const int ar  = tid >> 1;
    const int ak  = (tid & 1) * 16;
    const int agr = min(rowBase + ar, nRows - 1);
    const float* aRow = A + (long)agr * lda + ak;

    // W staging: thread -> (k row, 16-wide n chunk); never out of bounds.
    const int bk = tid >> 3;
    const int bn = (tid & 7) * 16;
    const float* wRow0 = W + (long)bk * ldw + colBase + bn;

    // 8 waves: 4 row groups x 2 col groups, each wave owns 32x64
    const int wR = (wid & 3) * 32;
    const int wC = (wid >> 2) * 64;

    v8f acc[2][4];
#pragma unroll
    for (int i = 0; i < 2; ++i)
#pragma unroll
        for (int j = 0; j < 4; ++j)
#pragma unroll
            for (int r = 0; r < 8; ++r)
                acc[i][j][r] = 0.0f;

    for (int k0 = 0; k0 < K; k0 += 32) {
        // ---- stage A tile (128x32): 4x b128 loads, pack, 2x b128 LDS stores ----
        {
            const float* ap = aRow + k0;
            const float4 f0 = *reinterpret_cast<const float4*>(ap + 0);
            const float4 f1 = *reinterpret_cast<const float4*>(ap + 4);
            const float4 f2 = *reinterpret_cast<const float4*>(ap + 8);
            const float4 f3 = *reinterpret_cast<const float4*>(ap + 12);
            __builtin_prefetch(ap + 32, 0, 3);   // speculative: dropped if invalid
            uint4 u0, u1;
            u0.x = pack2bf(f0.x, f0.y);
            u0.y = pack2bf(f0.z, f0.w);
            u0.z = pack2bf(f1.x, f1.y);
            u0.w = pack2bf(f1.z, f1.w);
            u1.x = pack2bf(f2.x, f2.y);
            u1.y = pack2bf(f2.z, f2.w);
            u1.z = pack2bf(f3.x, f3.y);
            u1.w = pack2bf(f3.z, f3.w);
            uint4* dp = reinterpret_cast<uint4*>(&As[ar][ak]);
            dp[0] = u0;
            dp[1] = u1;
        }
        // ---- stage W tile (32x128) transposed into Bs[n][k]; no bounds needed ----
        {
            const float* wp = wRow0 + (long)k0 * ldw;
            const float4 w0 = *reinterpret_cast<const float4*>(wp + 0);
            const float4 w1 = *reinterpret_cast<const float4*>(wp + 4);
            const float4 w2 = *reinterpret_cast<const float4*>(wp + 8);
            const float4 w3 = *reinterpret_cast<const float4*>(wp + 12);
            Bs[bn +  0][bk] = f2bf(w0.x);
            Bs[bn +  1][bk] = f2bf(w0.y);
            Bs[bn +  2][bk] = f2bf(w0.z);
            Bs[bn +  3][bk] = f2bf(w0.w);
            Bs[bn +  4][bk] = f2bf(w1.x);
            Bs[bn +  5][bk] = f2bf(w1.y);
            Bs[bn +  6][bk] = f2bf(w1.z);
            Bs[bn +  7][bk] = f2bf(w1.w);
            Bs[bn +  8][bk] = f2bf(w2.x);
            Bs[bn +  9][bk] = f2bf(w2.y);
            Bs[bn + 10][bk] = f2bf(w2.z);
            Bs[bn + 11][bk] = f2bf(w2.w);
            Bs[bn + 12][bk] = f2bf(w3.x);
            Bs[bn + 13][bk] = f2bf(w3.y);
            Bs[bn + 14][bk] = f2bf(w3.z);
            Bs[bn + 15][bk] = f2bf(w3.w);
        }
        __syncthreads();

        // ---- fragment loads per ISA 7.12.2 16-bit layouts ----
        AFrag a[2], b[4];
#pragma unroll
        for (int s = 0; s < 2; ++s) {
            // A 16x32: lane(0-15)=row M, elems 0..7 -> k=laneHi*8+0..7,
            //          elems 8..15 -> k=laneHi*8+16..23
            const unsigned short* rp = &As[wR + s * 16 + lane15][0];
            a[s].q[0] = *reinterpret_cast<const uint4*>(rp + laneHi * 8);
            a[s].q[1] = *reinterpret_cast<const uint4*>(rp + laneHi * 8 + 16);
        }
#pragma unroll
        for (int s = 0; s < 4; ++s) {
            // B 32x16: lane(0-15)=col N, elems 0..7 -> k=laneHi*8+0..7, elems 8..15 -> k=laneHi*8+16..23
            const unsigned short* cp = &Bs[wC + s * 16 + lane15][0];
            b[s].q[0] = *reinterpret_cast<const uint4*>(cp + laneHi * 8);
            b[s].q[1] = *reinterpret_cast<const uint4*>(cp + laneHi * 8 + 16);
        }

#pragma unroll
        for (int i = 0; i < 2; ++i)
#pragma unroll
            for (int j = 0; j < 4; ++j)
                acc[i][j] = __builtin_amdgcn_wmma_f32_16x16x32_bf16(
                    false, a[i].v, false, b[j].v,
                    (short)0, acc[i][j], false, false);

        __syncthreads();
    }

    // ---- epilogue: VGPR r -> (M=r, N=lane) lanes 0-15; (M=r+8, N=lane-16) hi ----
#pragma unroll
    for (int i = 0; i < 2; ++i) {
#pragma unroll
        for (int j = 0; j < 4; ++j) {
            const int m0 = wR + i * 16 + laneHi * 8;
            const int gn = colBase + wC + j * 16 + lane15;
            const float bv = bias ? bias[gn] : 0.0f;
#pragma unroll
            for (int r = 0; r < 8; ++r) {
                const int gr = rowBase + m0 + r;
                if (fullRows || gr < nRows) {
                    float v = acc[i][j][r] + bv;
                    if (doAct) v = lrelu(v);
                    C[(long)gr * ldc + gn] = v;
                }
            }
        }
    }
}

// ---------------------------------------------------------------------------
// Small utility kernels
// ---------------------------------------------------------------------------
__global__ void fill_zero_kernel(float* p, long n) {
    long i = (long)blockIdx.x * blockDim.x + threadIdx.x;
    long s = (long)gridDim.x * blockDim.x;
    for (; i < n; i += s) p[i] = 0.0f;
}

__global__ void copy_kernel(float* __restrict__ d, const float* __restrict__ s, long n) {
    long i  = (long)blockIdx.x * blockDim.x + threadIdx.x;
    long st = (long)gridDim.x * blockDim.x;
    for (; i < n; i += st) d[i] = s[i];
}

// pad x_ft (N x 54) -> (N x 64) with zero columns 54..63; one wave per row
__global__ void pad_xft_kernel(const float* __restrict__ src,
                               float* __restrict__ dst, int nRows) {
    int gid  = blockIdx.x * blockDim.x + threadIdx.x;
    int i    = gid >> 5;
    int lane = gid & 31;
    if (i >= nRows) return;
    int c = lane * 2;
    float2 o;
    o.x = (c     < 54) ? src[(long)i * 54 + c]     : 0.0f;
    o.y = (c + 1 < 54) ? src[(long)i * 54 + c + 1] : 0.0f;
    *reinterpret_cast<float2*>(dst + (long)i * 64 + c) = o;
}

__global__ void bias_sum3_kernel(const float* a, const float* b, const float* c,
                                 float* o, int n) {
    int i = blockIdx.x * blockDim.x + threadIdx.x;
    if (i < n) o[i] = a[i] + b[i] + c[i];
}

__global__ void bias_sum4_kernel(const float* a, const float* b, const float* c,
                                 const float* d, float* o, int n) {
    int i = blockIdx.x * blockDim.x + threadIdx.x;
    if (i < n) o[i] = a[i] + b[i] + c[i] + d[i];
}

__global__ void deg_count_kernel(const int* __restrict__ dst, float* __restrict__ deg, int E_) {
    int e = blockIdx.x * blockDim.x + threadIdx.x;
    if (e < E_) atomicAdd(&deg[dst[e]], 1.0f);
}

__global__ void deg_inv_kernel(float* deg, int n) {
    int i = blockIdx.x * blockDim.x + threadIdx.x;
    if (i < n) deg[i] = rsqrtf(deg[i] + 2.0f);
}

// out[i, 0:128] = hW[i,:] * (2*dinv^2) + bias    (one wave32 per node, 4 cols/lane)
__global__ void gcn_init_kernel(const float* __restrict__ hW,
                                const float* __restrict__ dinv,
                                const float* __restrict__ bias,
                                float* __restrict__ out, int ldo, int nRows)
{
    int gid  = blockIdx.x * blockDim.x + threadIdx.x;
    int i    = gid >> 5;
    int lane = gid & 31;
    if (i >= nRows) return;
    float di = dinv[i];
    float sc = 2.0f * di * di;
    const float4 v = *reinterpret_cast<const float4*>(hW + (long)i * 128 + lane * 4);
    const float4 b = *reinterpret_cast<const float4*>(bias + lane * 4);
    float4 o;
    o.x = v.x * sc + b.x;  o.y = v.y * sc + b.y;
    o.z = v.z * sc + b.z;  o.w = v.w * sc + b.w;
    *reinterpret_cast<float4*>(out + (long)i * ldo + lane * 4) = o;
}

// one wave32 per edge: out[dst,:128] += hW[src,:128] * (dinv[src]*dinv[dst])
__global__ void gcn_scatter_kernel(const float* __restrict__ hW,
                                   const int* __restrict__ src,
                                   const int* __restrict__ dst,
                                   const float* __restrict__ dinv,
                                   float* __restrict__ out, int ldo, int E_)
{
    int gid  = blockIdx.x * blockDim.x + threadIdx.x;
    int e    = gid >> 5;
    int lane = gid & 31;
    if (e >= E_) return;
    int s = src[e], d = dst[e];
    float coef = dinv[s] * dinv[d];
    const float4 v = *reinterpret_cast<const float4*>(hW + (long)s * 128 + lane * 4);
    float* op = out + (long)d * ldo + lane * 4;
    atomicAdd(op + 0, v.x * coef);
    atomicAdd(op + 1, v.y * coef);
    atomicAdd(op + 2, v.z * coef);
    atomicAdd(op + 3, v.w * coef);
}

__global__ void lrelu_slice_kernel(float* out, int ldo, int nRows) {
    int gid  = blockIdx.x * blockDim.x + threadIdx.x;
    int i    = gid >> 5;
    int lane = gid & 31;
    if (i >= nRows) return;
    float4* p = reinterpret_cast<float4*>(out + (long)i * ldo + lane * 4);
    float4 v = *p;
    v.x = lrelu(v.x); v.y = lrelu(v.y); v.z = lrelu(v.z); v.w = lrelu(v.w);
    *p = v;
}

// one wave32 per node: out[i,0:2] = drug_h[i]@W_od + b_od ; out[i,2:4] = dis path
__global__ void head_kernel(const float* __restrict__ drug_h,
                            const float* __restrict__ dis_h,
                            const float* __restrict__ Wod, const float* __restrict__ bod,
                            const float* __restrict__ Wods, const float* __restrict__ bods,
                            float* __restrict__ out, int nRows)
{
    int gid  = blockIdx.x * blockDim.x + threadIdx.x;
    int i    = gid >> 5;
    int lane = gid & 31;
    if (i >= nRows) return;
    const float4 dh = *reinterpret_cast<const float4*>(drug_h + (long)i * 128 + lane * 4);
    const float4 sh = *reinterpret_cast<const float4*>(dis_h  + (long)i * 128 + lane * 4);
    int k = lane * 4;
    float s0 = dh.x * Wod[(k + 0) * 2 + 0] + dh.y * Wod[(k + 1) * 2 + 0]
             + dh.z * Wod[(k + 2) * 2 + 0] + dh.w * Wod[(k + 3) * 2 + 0];
    float s1 = dh.x * Wod[(k + 0) * 2 + 1] + dh.y * Wod[(k + 1) * 2 + 1]
             + dh.z * Wod[(k + 2) * 2 + 1] + dh.w * Wod[(k + 3) * 2 + 1];
    float t0 = sh.x * Wods[(k + 0) * 2 + 0] + sh.y * Wods[(k + 1) * 2 + 0]
             + sh.z * Wods[(k + 2) * 2 + 0] + sh.w * Wods[(k + 3) * 2 + 0];
    float t1 = sh.x * Wods[(k + 0) * 2 + 1] + sh.y * Wods[(k + 1) * 2 + 1]
             + sh.z * Wods[(k + 2) * 2 + 1] + sh.w * Wods[(k + 3) * 2 + 1];
    for (int off = 16; off > 0; off >>= 1) {
        s0 += __shfl_down(s0, off);
        s1 += __shfl_down(s1, off);
        t0 += __shfl_down(t0, off);
        t1 += __shfl_down(t1, off);
    }
    if (lane == 0) {
        out[(long)i * 4 + 0] = s0 + bod[0];
        out[(long)i * 4 + 1] = s1 + bod[1];
        out[(long)i * 4 + 2] = t0 + bods[0];
        out[(long)i * 4 + 3] = t1 + bods[1];
    }
}

// ---------------------------------------------------------------------------
// Launcher
// ---------------------------------------------------------------------------
static inline void launch_gemm(const float* A, int lda,
                               const float* W, const float* bias,
                               float* C, int ldc,
                               int nRows, int K, int M, int doAct,
                               hipStream_t stream)
{
    dim3 grid((M + 127) / 128, (nRows + 127) / 128);
    gemm_bias_act_kernel<<<grid, 256, 0, stream>>>(A, lda, W, M, bias,
                                                   C, ldc, nRows, K, M, doAct);
}

extern "C" void kernel_launch(void* const* d_in, const int* in_sizes, int n_in,
                              void* d_out, int out_size, void* d_ws, size_t ws_size,
                              hipStream_t stream)
{
    (void)n_in; (void)out_size; (void)ws_size;

    const float* x    = (const float*)d_in[0];
    const float* x_ft = (const float*)d_in[1];
    const int*   ei   = (const int*)  d_in[2];

    const float* W_pe  = (const float*)d_in[3];  const float* b_pe  = (const float*)d_in[4];
    const float* W_gos = (const float*)d_in[5];  const float* b_gos = (const float*)d_in[6];
    const float* W_prot= (const float*)d_in[7];  const float* b_prot= (const float*)d_in[8];
    const float* W_om0 = (const float*)d_in[9];  const float* b_om0 = (const float*)d_in[10];
    const float* W_om1 = (const float*)d_in[11]; const float* b_om1 = (const float*)d_in[12];
    const float* W_om2 = (const float*)d_in[13]; const float* b_om2 = (const float*)d_in[14];
    const float* W_om3 = (const float*)d_in[15]; const float* b_om3 = (const float*)d_in[16];
    const float* W_c[3]  = {(const float*)d_in[17], (const float*)d_in[21], (const float*)d_in[25]};
    const float* b_c[3]  = {(const float*)d_in[18], (const float*)d_in[22], (const float*)d_in[26]};
    const float* W_o[3]  = {(const float*)d_in[19], (const float*)d_in[23], (const float*)d_in[27]};
    const float* b_o[3]  = {(const float*)d_in[20], (const float*)d_in[24], (const float*)d_in[28]};
    const float* W_lat = (const float*)d_in[29]; const float* b_lat = (const float*)d_in[30];
    const float* W_drug= (const float*)d_in[31]; const float* b_drug= (const float*)d_in[32];
    const float* W_dis = (const float*)d_in[33]; const float* b_dis = (const float*)d_in[34];
    const float* W_odr = (const float*)d_in[35]; const float* b_odr = (const float*)d_in[36];
    const float* W_odi = (const float*)d_in[37]; const float* b_odi = (const float*)d_in[38];

    const int FIN = 128 + 128 + 480;   // 736
    const int D   = 384;
    const int H   = 128;
    const int N_  = in_sizes[0] / FIN;
    const int E_  = in_sizes[2] / 2;
    const int* src = ei;
    const int* dst = ei + E_;

    // ---- workspace layout (floats), 256-elem aligned; z aliases h0, heads alias g0 ----
    float* ws = (float*)d_ws;
    size_t off = 0;
    auto alloc = [&](size_t n) { size_t o = off; off += (n + 255) & ~(size_t)255; return ws + o; };
    float* deg    = alloc((size_t)N_);
    float* wcat   = alloc((size_t)FIN * D);        // 736x384
    float* womcat = alloc((size_t)64 * D);         // 64x384 (K padded 54->64)
    float* bias1  = alloc(D);
    float* bias2  = alloc(D);
    float* xft64  = alloc((size_t)N_ * 64);        // x_ft zero-padded 54->64 cols
    float* h0z    = alloc((size_t)N_ * D);         // h0, later z
    float* g0d    = alloc((size_t)N_ * D);         // g0, later drug_h | dis_h
    float* hW     = alloc((size_t)N_ * H);
    float* jk     = alloc((size_t)N_ * 2 * D);     // N x 768: [hs | gs]

    const int nWaveBlocksN = (N_ * 32 + 255) / 256;
    const int nWaveBlocksE = (E_ * 32 + 255) / 256;

    // ---- degree / deg_inv ----
    fill_zero_kernel<<<1024, 256, 0, stream>>>(deg, (long)N_);
    deg_count_kernel<<<(E_ + 255) / 256, 256, 0, stream>>>(dst, deg, E_);
    deg_inv_kernel<<<(N_ + 255) / 256, 256, 0, stream>>>(deg, N_);

    // ---- pack weights + pad omics features ----
    copy_kernel<<<512, 256, 0, stream>>>(wcat,             W_pe,  (long)128 * D);
    copy_kernel<<<512, 256, 0, stream>>>(wcat + 128 * D,   W_gos, (long)128 * D);
    copy_kernel<<<512, 256, 0, stream>>>(wcat + 256 * D,   W_prot,(long)480 * D);
    fill_zero_kernel<<<256, 256, 0, stream>>>(womcat, (long)64 * D);
    copy_kernel<<<64, 256, 0, stream>>>(womcat,            W_om0, (long)3  * D);
    copy_kernel<<<64, 256, 0, stream>>>(womcat + 3  * D,   W_om1, (long)3  * D);
    copy_kernel<<<64, 256, 0, stream>>>(womcat + 6  * D,   W_om2, (long)24 * D);
    copy_kernel<<<64, 256, 0, stream>>>(womcat + 30 * D,   W_om3, (long)24 * D);
    bias_sum3_kernel<<<2, 256, 0, stream>>>(b_pe, b_gos, b_prot, bias1, D);
    bias_sum4_kernel<<<2, 256, 0, stream>>>(b_om0, b_om1, b_om2, b_om3, bias2, D);
    pad_xft_kernel<<<nWaveBlocksN, 256, 0, stream>>>(x_ft, xft64, N_);

    // ---- input embeddings (WMMA GEMMs) ----
    launch_gemm(x,     FIN, wcat,   bias1, h0z, D, N_, FIN, D, 1, stream);  // h0
    launch_gemm(xft64, 64,  womcat, bias2, g0d, D, N_, 64,  D, 1, stream);  // g0

    // ---- 3 GCN layers, both branches; outputs land in jk column slices ----
    const int ldjk = 2 * D;   // 768
    for (int j = 0; j < 3; ++j) {
        const int Kj = (j == 0) ? D : H;
        // c-branch
        {
            const float* Ain = (j == 0) ? h0z : (jk + (size_t)(j - 1) * H);
            const int    lda = (j == 0) ? D : ldjk;
            launch_gemm(Ain, lda, W_c[j], nullptr, hW, H, N_, Kj, H, 0, stream);
            float* slice = jk + (size_t)j * H;
            gcn_init_kernel<<<nWaveBlocksN, 256, 0, stream>>>(hW, deg, b_c[j], slice, ldjk, N_);
            gcn_scatter_kernel<<<nWaveBlocksE, 256, 0, stream>>>(hW, src, dst, deg, slice, ldjk, E_);
            lrelu_slice_kernel<<<nWaveBlocksN, 256, 0, stream>>>(slice, ldjk, N_);
        }
        // o-branch
        {
            const float* Ain = (j == 0) ? g0d : (jk + (size_t)D + (size_t)(j - 1) * H);
            const int    lda = (j == 0) ? D : ldjk;
            launch_gemm(Ain, lda, W_o[j], nullptr, hW, H, N_, Kj, H, 0, stream);
            float* slice = jk + (size_t)D + (size_t)j * H;
            gcn_init_kernel<<<nWaveBlocksN, 256, 0, stream>>>(hW, deg, b_o[j], slice, ldjk, N_);
            gcn_scatter_kernel<<<nWaveBlocksE, 256, 0, stream>>>(hW, src, dst, deg, slice, ldjk, E_);
            lrelu_slice_kernel<<<nWaveBlocksN, 256, 0, stream>>>(slice, ldjk, N_);
        }
    }

    // ---- latent fusion z = act(jk @ W_lat + b_lat)  (z aliases h0 buffer) ----
    launch_gemm(jk, ldjk, W_lat, b_lat, h0z, D, N_, ldjk, D, 1, stream);

    // ---- heads: hidden layers via WMMA, final 128->2 via wave reduction ----
    float* drug_h = g0d;
    float* dis_h  = g0d + (size_t)N_ * H;
    launch_gemm(h0z, D, W_drug, b_drug, drug_h, H, N_, D, H, 1, stream);
    launch_gemm(h0z, D, W_dis,  b_dis,  dis_h,  H, N_, D, H, 1, stream);
    head_kernel<<<nWaveBlocksN, 256, 0, stream>>>(drug_h, dis_h, W_odr, b_odr,
                                                  W_odi, b_odi, (float*)d_out, N_);
}
